// HierarchicalRNAEncoder_71270687310556
// MI455X (gfx1250) — matrix operand order repaired and checked by reading, compiled, not run on hardware
//
#include <hip/hip_runtime.h>
#include <cstdint>
#include <cstddef>

typedef __bf16 bf16_t;
typedef bf16_t v16bf __attribute__((ext_vector_type(16)));
typedef float  v8f   __attribute__((ext_vector_type(8)));

namespace {

constexpr int kL  = 6;
constexpr int kD  = 640;
constexpr int kB  = 8;
constexpr int kS  = 512;
constexpr int kM  = kB * kS;   // 4096 token rows
constexpr int kFF = 2048;
constexpr int kPE = 256, kSE = 256, kTE = 128;
constexpr float kPairBonus = 2.0f;

struct alignas(16) U128 { unsigned int x[4]; };
union FragBits { U128 q[2]; v16bf v; };

// A fragment (16x32 bf16, MxK): lane m holds row m; half 0 -> K k0..k0+15,
// half 1 -> K k0+16..k0+31 (matches frag_B). Two 16B loads per lane.
__device__ inline v16bf frag_A(const bf16_t* __restrict__ base, int ld, int row, int k0, int lane) {
  const int m = lane & 15, half = lane >> 4;
  const bf16_t* p = base + (size_t)(row + m) * ld + k0 + half * 16;
  FragBits u;
  u.q[0] = *reinterpret_cast<const U128*>(p);
  u.q[1] = *reinterpret_cast<const U128*>(p + 8);
  return u.v;
}

// B fragment (32x16 bf16, KxN) from row-major W[N,K]: lane n holds column n = W row n;
// half 0 -> K k0..k0+15, half 1 -> K k0+16..k0+31. 32B contiguous per lane.
__device__ inline v16bf frag_B(const bf16_t* __restrict__ base, int ld, int col, int k0, int lane) {
  const int n = lane & 15, half = lane >> 4;
  const bf16_t* p = base + (size_t)(col + n) * ld + k0 + half * 16;
  FragBits u;
  u.q[0] = *reinterpret_cast<const U128*>(p);
  u.q[1] = *reinterpret_cast<const U128*>(p + 8);
  return u.v;
}

// B fragment from row-major V[K,N] (stride ld between K rows): strided gather.
__device__ inline v16bf frag_Bk(const bf16_t* __restrict__ base, int ld, int col, int k0, int lane) {
  const int n = lane & 15, half = lane >> 4;
  v16bf v;
#pragma unroll
  for (int j = 0; j < 16; ++j)
    v[j] = base[(size_t)(k0 + half * 16 + j) * ld + col + n];
  return v;
}

__device__ inline v8f wmma_bf16(v16bf a, v16bf b, v8f c) {
  return __builtin_amdgcn_wmma_f32_16x16x32_bf16(false, a, false, b, (short)0, c, false, false);
}

__device__ inline v8f zero8() {
  v8f a;
#pragma unroll
  for (int i = 0; i < 8; ++i) a[i] = 0.0f;
  return a;
}

// ---------------- generic GEMM: C[M,N] = alpha*(A[M,K] @ W[N,K]^T) + bias (+resid)(ReLU) -------
// Block = 256 threads = 8 waves arranged 4(M) x 2(N); wave tile 32x64 (8 WMMA accumulators).
// Block tile 128x128. Per k-step: 6 fragment loads feed 8 WMMAs.
__global__ __launch_bounds__(256)
void gemm_wmma_kernel(const bf16_t* __restrict__ A, int lda,
                      const bf16_t* __restrict__ W, int ldw,
                      const float* __restrict__ bias,
                      const float* __restrict__ resid,
                      float* __restrict__ outF, bf16_t* __restrict__ outB,
                      int ldo, int K, float alpha, int relu) {
  const int lane = threadIdx.x & 31;
  const int wave = threadIdx.x >> 5;
  const int tm = blockIdx.y * 128 + (wave >> 1) * 32;
  const int tn = blockIdx.x * 128 + (wave & 1) * 64;
  v8f acc[2][4];
#pragma unroll
  for (int mi = 0; mi < 2; ++mi)
#pragma unroll
    for (int ni = 0; ni < 4; ++ni) acc[mi][ni] = zero8();

  for (int k0 = 0; k0 < K; k0 += 32) {
    v16bf a0 = frag_A(A, lda, tm,      k0, lane);
    v16bf a1 = frag_A(A, lda, tm + 16, k0, lane);
    v16bf b0 = frag_B(W, ldw, tn,      k0, lane);
    v16bf b1 = frag_B(W, ldw, tn + 16, k0, lane);
    v16bf b2 = frag_B(W, ldw, tn + 32, k0, lane);
    v16bf b3 = frag_B(W, ldw, tn + 48, k0, lane);
    acc[0][0] = wmma_bf16(a0, b0, acc[0][0]);
    acc[0][1] = wmma_bf16(a0, b1, acc[0][1]);
    acc[0][2] = wmma_bf16(a0, b2, acc[0][2]);
    acc[0][3] = wmma_bf16(a0, b3, acc[0][3]);
    acc[1][0] = wmma_bf16(a1, b0, acc[1][0]);
    acc[1][1] = wmma_bf16(a1, b1, acc[1][1]);
    acc[1][2] = wmma_bf16(a1, b2, acc[1][2]);
    acc[1][3] = wmma_bf16(a1, b3, acc[1][3]);
  }

  const int n = lane & 15, half = lane >> 4;
#pragma unroll
  for (int ni = 0; ni < 4; ++ni) {
    const int gc = tn + ni * 16 + n;
    const float bv = bias ? bias[gc] : 0.0f;
#pragma unroll
    for (int mi = 0; mi < 2; ++mi) {
#pragma unroll
      for (int r = 0; r < 8; ++r) {
        const int gr = tm + mi * 16 + r + half * 8;
        float v = acc[mi][ni][r] * alpha + bv;
        if (resid) v += resid[(size_t)gr * ldo + gc];
        if (relu)  v = fmaxf(v, 0.0f);
        if (outF)  outF[(size_t)gr * ldo + gc] = v;
        if (outB)  outB[(size_t)gr * ldo + gc] = (bf16_t)v;
      }
    }
  }
}

// ---------------- attention: scores = 0.125 * Q K^T per (b,h) ----------------
// Wave tile 32x64, block tile 128x128 over the 512x512 score matrix; K = 64 (2 k-steps).
__global__ __launch_bounds__(256)
void attn_qk_kernel(const bf16_t* __restrict__ qkv, int ld, int E, int nh,
                    float* __restrict__ scores, float alpha) {
  const int lane = threadIdx.x & 31;
  const int wave = threadIdx.x >> 5;
  const int z = blockIdx.z;
  const int b = z / nh, h = z % nh;
  const bf16_t* Q  = qkv + (size_t)b * kS * ld + h * 64;
  const bf16_t* Kp = Q + E;
  const int tm = blockIdx.y * 128 + (wave >> 1) * 32;
  const int tn = blockIdx.x * 128 + (wave & 1) * 64;
  v8f acc[2][4];
#pragma unroll
  for (int mi = 0; mi < 2; ++mi)
#pragma unroll
    for (int ni = 0; ni < 4; ++ni) acc[mi][ni] = zero8();

#pragma unroll
  for (int k0 = 0; k0 < 64; k0 += 32) {
    v16bf a0 = frag_A(Q, ld, tm,      k0, lane);
    v16bf a1 = frag_A(Q, ld, tm + 16, k0, lane);
    v16bf b0 = frag_B(Kp, ld, tn,      k0, lane);
    v16bf b1 = frag_B(Kp, ld, tn + 16, k0, lane);
    v16bf b2 = frag_B(Kp, ld, tn + 32, k0, lane);
    v16bf b3 = frag_B(Kp, ld, tn + 48, k0, lane);
    acc[0][0] = wmma_bf16(a0, b0, acc[0][0]);
    acc[0][1] = wmma_bf16(a0, b1, acc[0][1]);
    acc[0][2] = wmma_bf16(a0, b2, acc[0][2]);
    acc[0][3] = wmma_bf16(a0, b3, acc[0][3]);
    acc[1][0] = wmma_bf16(a1, b0, acc[1][0]);
    acc[1][1] = wmma_bf16(a1, b1, acc[1][1]);
    acc[1][2] = wmma_bf16(a1, b2, acc[1][2]);
    acc[1][3] = wmma_bf16(a1, b3, acc[1][3]);
  }

  float* C = scores + (size_t)z * kS * kS;
  const int n = lane & 15, half = lane >> 4;
#pragma unroll
  for (int mi = 0; mi < 2; ++mi)
#pragma unroll
    for (int ni = 0; ni < 4; ++ni)
#pragma unroll
      for (int r = 0; r < 8; ++r)
        C[(size_t)(tm + mi * 16 + r + half * 8) * kS + tn + ni * 16 + n] = acc[mi][ni][r] * alpha;
}

// ---------------- attention: ctx = P V per (b,h), stored [b*S+q, h*64+d] ----------------
// Wave tile 32x32, block tile 128x64 (full head dim); K = 512.
__global__ __launch_bounds__(256)
void attn_pv_kernel(const bf16_t* __restrict__ probs, const bf16_t* __restrict__ qkv,
                    int ld, int E, int nh, bf16_t* __restrict__ ctx) {
  const int lane = threadIdx.x & 31;
  const int wave = threadIdx.x >> 5;
  const int z = blockIdx.z;
  const int b = z / nh, h = z % nh;
  const bf16_t* P = probs + (size_t)z * kS * kS;
  const bf16_t* V = qkv + (size_t)b * kS * ld + 2 * E + h * 64;
  const int tm = blockIdx.y * 128 + (wave >> 1) * 32;
  const int tn = (wave & 1) * 32;
  v8f acc[2][2];
#pragma unroll
  for (int mi = 0; mi < 2; ++mi)
#pragma unroll
    for (int ni = 0; ni < 2; ++ni) acc[mi][ni] = zero8();

  for (int k0 = 0; k0 < kS; k0 += 32) {
    v16bf a0 = frag_A(P, kS, tm,      k0, lane);
    v16bf a1 = frag_A(P, kS, tm + 16, k0, lane);
    v16bf b0 = frag_Bk(V, ld, tn,      k0, lane);
    v16bf b1 = frag_Bk(V, ld, tn + 16, k0, lane);
    acc[0][0] = wmma_bf16(a0, b0, acc[0][0]);
    acc[0][1] = wmma_bf16(a0, b1, acc[0][1]);
    acc[1][0] = wmma_bf16(a1, b0, acc[1][0]);
    acc[1][1] = wmma_bf16(a1, b1, acc[1][1]);
  }

  const int n = lane & 15, half = lane >> 4;
#pragma unroll
  for (int mi = 0; mi < 2; ++mi)
#pragma unroll
    for (int ni = 0; ni < 2; ++ni)
#pragma unroll
      for (int r = 0; r < 8; ++r) {
        const int q = tm + mi * 16 + r + half * 8;
        ctx[(size_t)(b * kS + q) * E + h * 64 + tn + ni * 16 + n] = (bf16_t)acc[mi][ni][r];
      }
}

// ---------------- softmax over keys (row length 512), + pairing bonus, key padding mask --------
__global__ __launch_bounds__(256)
void softmax_kernel(const float* __restrict__ scores, bf16_t* __restrict__ probs,
                    const float* __restrict__ pmask, const unsigned char* __restrict__ kpm,
                    int nh, float bonus) {
  __shared__ float red[256];
  const int q = blockIdx.x;
  const int z = blockIdx.y;
  const int b = z / nh;
  const int t = threadIdx.x;
  const float* srow = scores + (size_t)z * kS * kS + (size_t)q * kS;
  float e0 = srow[t], e1 = srow[t + 256];
  if (pmask) {
    const float* pr = pmask + ((size_t)b * kS + q) * kS;
    e0 += pr[t] * bonus;
    e1 += pr[t + 256] * bonus;
  }
  if (kpm) {
    const unsigned char* kr = kpm + (size_t)b * kS;
    if (kr[t])       e0 = -1e30f;
    if (kr[t + 256]) e1 = -1e30f;
  }
  red[t] = fmaxf(e0, e1);
  __syncthreads();
  for (int s = 128; s > 0; s >>= 1) { if (t < s) red[t] = fmaxf(red[t], red[t + s]); __syncthreads(); }
  const float mx = red[0];
  __syncthreads();
  const float x0 = __expf(e0 - mx), x1 = __expf(e1 - mx);
  red[t] = x0 + x1;
  __syncthreads();
  for (int s = 128; s > 0; s >>= 1) { if (t < s) red[t] += red[t + s]; __syncthreads(); }
  const float inv = 1.0f / red[0];
  bf16_t* prow = probs + (size_t)z * kS * kS + (size_t)q * kS;
  prow[t]       = (bf16_t)(x0 * inv);
  prow[t + 256] = (bf16_t)(x1 * inv);
}

// ---------------- LayerNorm over D=640, f32 in -> bf16 out ----------------
__global__ __launch_bounds__(256)
void layernorm_kernel(const float* __restrict__ x, const float* __restrict__ g,
                      const float* __restrict__ bta, bf16_t* __restrict__ out) {
  __shared__ float red[256];
  const int row = blockIdx.x;
  const int t = threadIdx.x;
  const float* xr = x + (size_t)row * kD;
  const float v0 = xr[t];
  const float v1 = xr[t + 256];
  const float v2 = (t < kD - 512) ? xr[t + 512] : 0.0f;
  red[t] = v0 + v1 + v2;
  __syncthreads();
  for (int s = 128; s > 0; s >>= 1) { if (t < s) red[t] += red[t + s]; __syncthreads(); }
  const float mean = red[0] * (1.0f / kD);
  __syncthreads();
  const float d0 = v0 - mean, d1 = v1 - mean;
  const float d2 = (t < kD - 512) ? (v2 - mean) : 0.0f;
  red[t] = d0 * d0 + d1 * d1 + d2 * d2;
  __syncthreads();
  for (int s = 128; s > 0; s >>= 1) { if (t < s) red[t] += red[t + s]; __syncthreads(); }
  const float rstd = rsqrtf(red[0] * (1.0f / kD) + 1e-5f);
  bf16_t* orow = out + (size_t)row * kD;
  orow[t]       = (bf16_t)(d0 * rstd * g[t] + bta[t]);
  orow[t + 256] = (bf16_t)(d1 * rstd * g[t + 256] + bta[t + 256]);
  if (t < kD - 512)
    orow[t + 512] = (bf16_t)(d2 * rstd * g[t + 512] + bta[t + 512]);
}

__global__ __launch_bounds__(256)
void convert_kernel(const float* __restrict__ in, bf16_t* __restrict__ out, size_t n) {
  const size_t i = (size_t)blockIdx.x * 256 + threadIdx.x;
  if (i < n) out[i] = (bf16_t)in[i];
}

__global__ __launch_bounds__(256)
void copy_kernel(const float* __restrict__ in, float* __restrict__ out, size_t n) {
  const size_t i = (size_t)blockIdx.x * 256 + threadIdx.x;
  if (i < n) out[i] = in[i];
}

inline void gemm(hipStream_t st, const bf16_t* A, int lda, const bf16_t* W, int ldw,
                 const float* bias, const float* resid, float* outF, bf16_t* outB,
                 int ldo, int M, int N, int K, float alpha, int relu) {
  dim3 g(N / 128, M / 128, 1);
  gemm_wmma_kernel<<<g, 256, 0, st>>>(A, lda, W, ldw, bias, resid, outF, outB, ldo, K, alpha, relu);
}

} // namespace

extern "C" void kernel_launch(void* const* d_in, const int* in_sizes, int n_in,
                              void* d_out, int out_size, void* d_ws, size_t ws_size,
                              hipStream_t stream) {
  (void)in_sizes; (void)n_in; (void)out_size; (void)ws_size;

  // ---- workspace layout (bump allocator, 256B aligned) ----
  char* base = (char*)d_ws;
  size_t off = 0;
  auto take = [&](size_t bytes) -> char* {
    char* p = base + off;
    off += (bytes + 255) & ~(size_t)255;
    return p;
  };
  float*  xA     = (float*)take((size_t)kM * kD * 4);
  float*  xB     = (float*)take((size_t)kM * kD * 4);
  bf16_t* xnb    = (bf16_t*)take((size_t)kM * kD * 2);
  bf16_t* ppb    = (bf16_t*)take((size_t)kM * kPE * 2);
  bf16_t* spb    = (bf16_t*)take((size_t)kM * kSE * 2);
  bf16_t* tpb    = (bf16_t*)take((size_t)kM * kTE * 2);
  bf16_t* qkvb   = (bf16_t*)take((size_t)kM * 3 * kPE * 2);
  float*  scores = (float*)take((size_t)kB * 4 * kS * kS * 4);
  bf16_t* probs  = (bf16_t*)take((size_t)kB * 4 * kS * kS * 2);
  bf16_t* ctx    = (bf16_t*)take((size_t)kM * kPE * 2);
  bf16_t* comb   = (bf16_t*)take((size_t)kM * kD * 2);
  bf16_t* hb     = (bf16_t*)take((size_t)kM * kFF * 2);
  bf16_t* wpin   = (bf16_t*)take((size_t)kL * kPE * kD * 2);
  bf16_t* wsin   = (bf16_t*)take((size_t)kL * kSE * kD * 2);
  bf16_t* wtin   = (bf16_t*)take((size_t)kL * kTE * kD * 2);
  bf16_t* wpqkv  = (bf16_t*)take((size_t)kL * 3 * kPE * kPE * 2);
  bf16_t* wpout  = (bf16_t*)take((size_t)kL * kPE * kPE * 2);
  bf16_t* wsqkv  = (bf16_t*)take((size_t)kL * 3 * kSE * kSE * 2);
  bf16_t* wsout  = (bf16_t*)take((size_t)kL * kSE * kSE * 2);
  bf16_t* wtqkv  = (bf16_t*)take((size_t)kL * 3 * kTE * kTE * 2);
  bf16_t* wtout  = (bf16_t*)take((size_t)kL * kTE * kTE * 2);
  bf16_t* wo     = (bf16_t*)take((size_t)kL * kD * kD * 2);
  bf16_t* wff1   = (bf16_t*)take((size_t)kL * kFF * kD * 2);
  bf16_t* wff2   = (bf16_t*)take((size_t)kL * kD * kFF * 2);

  const float* pairing_mask = (const float*)d_in[1];
  const unsigned char* kpm  = (const unsigned char*)d_in[2];

  // ---- bf16 weight cache (recomputed every call: deterministic, no state) ----
  auto conv = [&](const void* src, bf16_t* dst, size_t n) {
    convert_kernel<<<dim3((unsigned)((n + 255) / 256)), 256, 0, stream>>>((const float*)src, dst, n);
  };
  conv(d_in[5],  wpin,  (size_t)kL * kPE * kD);
  conv(d_in[7],  wsin,  (size_t)kL * kSE * kD);
  conv(d_in[9],  wtin,  (size_t)kL * kTE * kD);
  conv(d_in[11], wpqkv, (size_t)kL * 3 * kPE * kPE);
  conv(d_in[13], wpout, (size_t)kL * kPE * kPE);
  conv(d_in[15], wsqkv, (size_t)kL * 3 * kSE * kSE);
  conv(d_in[17], wsout, (size_t)kL * kSE * kSE);
  conv(d_in[19], wtqkv, (size_t)kL * 3 * kTE * kTE);
  conv(d_in[21], wtout, (size_t)kL * kTE * kTE);
  conv(d_in[23], wo,    (size_t)kL * kD * kD);
  conv(d_in[25], wff1,  (size_t)kL * kFF * kD);
  conv(d_in[27], wff2,  (size_t)kL * kD * kFF);

  // x = src
  copy_kernel<<<dim3((kM * kD + 255) / 256), 256, 0, stream>>>((const float*)d_in[0], xA, (size_t)kM * kD);

  for (int l = 0; l < kL; ++l) {
    const float* n1s   = (const float*)d_in[3]  + (size_t)l * kD;
    const float* n1b   = (const float*)d_in[4]  + (size_t)l * kD;
    const float* pinb  = (const float*)d_in[6]  + (size_t)l * kPE;
    const float* sinb  = (const float*)d_in[8]  + (size_t)l * kSE;
    const float* tinb  = (const float*)d_in[10] + (size_t)l * kTE;
    const float* pqkvb = (const float*)d_in[12] + (size_t)l * 3 * kPE;
    const float* poutb = (const float*)d_in[14] + (size_t)l * kPE;
    const float* sqkvb = (const float*)d_in[16] + (size_t)l * 3 * kSE;
    const float* soutb = (const float*)d_in[18] + (size_t)l * kSE;
    const float* tqkvb = (const float*)d_in[20] + (size_t)l * 3 * kTE;
    const float* toutb = (const float*)d_in[22] + (size_t)l * kTE;
    const float* ob    = (const float*)d_in[24] + (size_t)l * kD;
    const float* ff1b  = (const float*)d_in[26] + (size_t)l * kFF;
    const float* ff2b  = (const float*)d_in[28] + (size_t)l * kD;
    const float* n2s   = (const float*)d_in[29] + (size_t)l * kD;
    const float* n2b   = (const float*)d_in[30] + (size_t)l * kD;

    // LN1 -> bf16
    layernorm_kernel<<<dim3(kM), 256, 0, stream>>>(xA, n1s, n1b, xnb);

    // input projections
    gemm(stream, xnb, kD, wpin + (size_t)l * kPE * kD, kD, pinb, nullptr, nullptr, ppb, kPE, kM, kPE, kD, 1.0f, 0);
    gemm(stream, xnb, kD, wsin + (size_t)l * kSE * kD, kD, sinb, nullptr, nullptr, spb, kSE, kM, kSE, kD, 1.0f, 0);
    gemm(stream, xnb, kD, wtin + (size_t)l * kTE * kD, kD, tinb, nullptr, nullptr, tpb, kTE, kM, kTE, kD, 1.0f, 0);

    struct Br {
      int E, nh, colofs;
      const bf16_t *act, *qkvW, *outW;
      const float *qkvBias, *outBias, *pmask;
    };
    const Br brs[3] = {
      { kPE, 4, 0,           ppb, wpqkv + (size_t)l * 3 * kPE * kPE, wpout + (size_t)l * kPE * kPE, pqkvb, poutb, nullptr },
      { kSE, 4, kPE,         spb, wsqkv + (size_t)l * 3 * kSE * kSE, wsout + (size_t)l * kSE * kSE, sqkvb, soutb, pairing_mask },
      { kTE, 2, kPE + kSE,   tpb, wtqkv + (size_t)l * 3 * kTE * kTE, wtout + (size_t)l * kTE * kTE, tqkvb, toutb, nullptr },
    };

    for (int i = 0; i < 3; ++i) {
      const Br& r = brs[i];
      const int E3 = 3 * r.E;
      // QKV projection
      gemm(stream, r.act, r.E, r.qkvW, r.E, r.qkvBias, nullptr, nullptr, qkvb, E3, kM, E3, r.E, 1.0f, 0);
      // scores = Q K^T * (hd^-0.5), hd = 64
      dim3 gq(kS / 128, kS / 128, kB * r.nh);
      attn_qk_kernel<<<gq, 256, 0, stream>>>(qkvb, E3, r.E, r.nh, scores, 0.125f);
      // softmax (+ pairing bonus for stem branch, + key padding mask)
      dim3 gs(kS, kB * r.nh);
      softmax_kernel<<<gs, 256, 0, stream>>>(scores, probs, r.pmask, kpm, r.nh, kPairBonus);
      // ctx = P V, laid out [b*S+q, h*64+d]
      dim3 gp(1, kS / 128, kB * r.nh);
      attn_pv_kernel<<<gp, 256, 0, stream>>>(probs, qkvb, E3, r.E, r.nh, ctx);
      // out-projection, written into concat slot of comb
      gemm(stream, ctx, r.E, r.outW, r.E, r.outBias, nullptr, nullptr, comb + r.colofs, kD, kM, r.E, r.E, 1.0f, 0);
    }

    // x = x + comb @ o_W^T + o_b
    gemm(stream, comb, kD, wo + (size_t)l * kD * kD, kD, ob, xA, xB, nullptr, kD, kM, kD, kD, 1.0f, 0);
    // LN2 -> bf16
    layernorm_kernel<<<dim3(kM), 256, 0, stream>>>(xB, n2s, n2b, xnb);
    // h = relu(xn @ ff1^T + b) (bf16)
    gemm(stream, xnb, kD, wff1 + (size_t)l * kFF * kD, kD, ff1b, nullptr, nullptr, hb, kFF, kM, kFF, kD, 1.0f, 1);
    // x = x + h @ ff2^T + b  (last layer writes d_out)
    float* xdst = (l == kL - 1) ? (float*)d_out : xA;
    gemm(stream, hb, kFF, wff2 + (size_t)l * kD * kFF, kFF, ff2b, xB, xdst, nullptr, kD, kM, kD, kFF, 1.0f, 0);
  }
}
